// VaGNN_16320875724918
// MI455X (gfx1250) — matrix operand order repaired and checked by reading, not compiled, not run on hardware
//
#include <hip/hip_runtime.h>
#include <hip/hip_bf16.h>
#include <stdint.h>

#define N_LV0 100000
#define N_NET 50000
#define E_G   1200000
#define E_C   400000
#define D     64
#define H     128

typedef __attribute__((ext_vector_type(16))) __bf16 v16bf;
typedef __attribute__((ext_vector_type(8)))  float  v8f;
typedef __attribute__((__vector_size__(16))) int v4i;          // builtin's pointee type
typedef __attribute__((address_space(1))) v4i g_v4i;           // global (prints as __device__)
typedef __attribute__((address_space(3))) v4i l_v4i;           // LDS (prints as __shared__)

#if defined(__has_builtin)
#if __has_builtin(__builtin_amdgcn_global_load_async_to_lds_b128)
#define HAVE_ASYNC_LDS 1
#endif
#endif

__device__ __forceinline__ float leaky(float x) { return x >= 0.0f ? x : 0.01f * x; }

// Monotonic float<->uint encoding so segment-max can use integer atomicMax.
__device__ __forceinline__ unsigned enc_max(float f) {
    unsigned b = __float_as_uint(f);
    return (b & 0x80000000u) ? ~b : (b | 0x80000000u);
}
__device__ __forceinline__ float dec_max(unsigned u) {
    unsigned b = (u & 0x80000000u) ? (u ^ 0x80000000u) : ~u;
    if ((b & 0x7F800000u) == 0x7F800000u) return 0.0f;  // empty segment / non-finite -> 0
    return __uint_as_float(b);
}

// Stage 16KB (512 x v16bf) of fragment-major weights into LDS.
// Prefers the CDNA5 async global->LDS engine (ASYNCcnt) when available.
__device__ __forceinline__ void stage_weights_16k(const v16bf* __restrict__ g, v16bf* s) {
#ifdef HAVE_ASYNC_LDS
    {
        const uint4* gp = (const uint4*)g;  // 1024 x 16B
        uint4* sp = (uint4*)s;
#pragma unroll
        for (int it = 0; it < 4; ++it) {
            int idx = it * 256 + threadIdx.x;
            // Flat->LDS: low 32 bits of a flat LDS pointer are the LDS byte address
            // (ISA 10.2 aperture mapping); flat global address == global address.
            __builtin_amdgcn_global_load_async_to_lds_b128(
                (g_v4i*)(uintptr_t)(gp + idx),
                (l_v4i*)(uint32_t)(uintptr_t)(sp + idx),
                0, 0);
        }
#if __has_builtin(__builtin_amdgcn_s_wait_asynccnt)
        __builtin_amdgcn_s_wait_asynccnt(0);
#else
        asm volatile("s_wait_asynccnt 0" ::: "memory");
#endif
    }
#else
    for (int i = threadIdx.x; i < 512; i += 256) s[i] = g[i];
#endif
    __syncthreads();
}

__global__ void zero_u32(uint32_t* p, int n) {
    int t = blockIdx.x * blockDim.x + threadIdx.x;
    if (t < n) p[t] = 0u;
}

// Emit bf16 weights pre-swizzled into the exact WMMA B-fragment register image:
// frag[(kt*NT+nt)*32 + lane][i]  =  B[k = kt*32 + (lane>>4)*16 + i][n = nt*16 + (lane&15)]
// W0/W1: B = [Ws;Wn]^T (128x64, NT=4, KT=4).  WM: B = Wm0^T (64x128, NT=8, KT=2).
__global__ void prep_weights(const float* __restrict__ Ws0, const float* __restrict__ Wn0,
                             const float* __restrict__ Ws1, const float* __restrict__ Wn1,
                             const float* __restrict__ Wm0,
                             __bf16* __restrict__ W0f, __bf16* __restrict__ W1f,
                             __bf16* __restrict__ WMf) {
    int t = blockIdx.x * blockDim.x + threadIdx.x;
    if (t < 8192) {
        int i = t & 15, lane = (t >> 4) & 31, nt = (t >> 9) & 3, kt = (t >> 11) & 3;
        int k = kt * 32 + (lane >> 4) * 16 + i;
        int m = nt * 16 + (lane & 15);
        float v = (k < D) ? Ws0[m * D + k] : Wn0[m * D + (k - D)];
        W0f[t] = (__bf16)v;
    } else if (t < 16384) {
        int q = t - 8192;
        int i = q & 15, lane = (q >> 4) & 31, nt = (q >> 9) & 3, kt = (q >> 11) & 3;
        int k = kt * 32 + (lane >> 4) * 16 + i;
        int m = nt * 16 + (lane & 15);
        float v = (k < D) ? Ws1[m * D + k] : Wn1[m * D + (k - D)];
        W1f[q] = (__bf16)v;
    } else if (t < 24576) {
        int q = t - 16384;
        int i = q & 15, lane = (q >> 4) & 31, nt = (q >> 9) & 7, kt = (q >> 12) & 1;
        int k = kt * 32 + (lane >> 4) * 16 + i;
        int m = nt * 16 + (lane & 15);
        WMf[q] = (__bf16)Wm0[m * D + k];
    }
}

__global__ void degree_kernel(const int* __restrict__ dst, float* __restrict__ deg, int ne) {
    int t = blockIdx.x * blockDim.x + threadIdx.x;
    if (t < ne) atomicAdd(&deg[dst[t]], 1.0f);
}

// One thread per (edge, 32-feature half): gather row, atomic-add into msum (L2-resident).
__global__ void scatter_add(const float* __restrict__ x, const int* __restrict__ src,
                            const int* __restrict__ dst, float* __restrict__ msum, int ne) {
    int t = blockIdx.x * blockDim.x + threadIdx.x;
    int e = t >> 1;
    if (e >= ne) return;
    int h = (t & 1) * 32;
    const float4* xr = (const float4*)(x + (size_t)src[e] * D + h);
    float* o = msum + (size_t)dst[e] * D + h;
#pragma unroll
    for (int i = 0; i < 8; ++i) {
        float4 v = xr[i];
        atomicAdd(o + i * 4 + 0, v.x);
        atomicAdd(o + i * 4 + 1, v.y);
        atomicAdd(o + i * 4 + 2, v.z);
        atomicAdd(o + i * 4 + 3, v.w);
    }
}

// Segment-max over full 64-wide rows (both halves of the reference are segment_max).
__global__ void pool_max(const float* __restrict__ x, const int* __restrict__ csrc,
                         const int* __restrict__ cdst, unsigned* __restrict__ yenc, int ne) {
    int t = blockIdx.x * blockDim.x + threadIdx.x;
    int e = t >> 1;
    if (e >= ne) return;
    int h = (t & 1) * 32;
    const float4* xr = (const float4*)(x + (size_t)csrc[e] * D + h);
    unsigned* o = yenc + (size_t)cdst[e] * D + h;
#pragma unroll
    for (int i = 0; i < 8; ++i) {
        float4 v = xr[i];
        atomicMax(o + i * 4 + 0, enc_max(v.x));
        atomicMax(o + i * 4 + 1, enc_max(v.y));
        atomicMax(o + i * 4 + 2, enc_max(v.z));
        atomicMax(o + i * 4 + 3, enc_max(v.w));
    }
}

// SAGE layer: out = leaky([Xin | msum/deg] (Nx128) @ Wc^T (128x64) + bias)
// 8 waves/block, 16-row strip per wave, 4 col-tiles, K=128 in 4 steps of 32.
__global__ void __launch_bounds__(256)
sage_wmma(const float* __restrict__ Xin, const float* __restrict__ msum,
          const float* __restrict__ deg, const v16bf* __restrict__ Wfrag,
          const float* __restrict__ bias, float* __restrict__ outp, int nrows) {
    __shared__ v16bf sW[512];  // 16 fragments x 32 lanes, 16KB
    stage_weights_16k(Wfrag, sW);

    const int wave = threadIdx.x >> 5;
    const int lane = threadIdx.x & 31;
    const int m0 = blockIdx.x * 128 + wave * 16;
    if (m0 >= nrows) return;  // wave-uniform: EXEC stays all-ones for WMMA

    const int lrow = lane & 15;
    const int khalf = lane >> 4;
    const int m = m0 + lrow;
    const float rdeg = 1.0f / fmaxf(deg[m], 1.0f);
    const float* xrow = Xin + (size_t)m * D;
    const float* nrow = msum + (size_t)m * D;

    v8f acc[4] = {};
#pragma unroll
    for (int kt = 0; kt < 4; ++kt) {
        const float* srcrow = (kt < 2) ? xrow : nrow;
        const float scale = (kt < 2) ? 1.0f : rdeg;
        const int kb = (kt & 1) * 32 + khalf * 16;
        float4 p0 = *(const float4*)(srcrow + kb);
        float4 p1 = *(const float4*)(srcrow + kb + 4);
        float4 p2 = *(const float4*)(srcrow + kb + 8);
        float4 p3 = *(const float4*)(srcrow + kb + 12);
        v16bf a;
        a[0]  = (__bf16)(p0.x * scale); a[1]  = (__bf16)(p0.y * scale);
        a[2]  = (__bf16)(p0.z * scale); a[3]  = (__bf16)(p0.w * scale);
        a[4]  = (__bf16)(p1.x * scale); a[5]  = (__bf16)(p1.y * scale);
        a[6]  = (__bf16)(p1.z * scale); a[7]  = (__bf16)(p1.w * scale);
        a[8]  = (__bf16)(p2.x * scale); a[9]  = (__bf16)(p2.y * scale);
        a[10] = (__bf16)(p2.z * scale); a[11] = (__bf16)(p2.w * scale);
        a[12] = (__bf16)(p3.x * scale); a[13] = (__bf16)(p3.y * scale);
        a[14] = (__bf16)(p3.z * scale); a[15] = (__bf16)(p3.w * scale);
#pragma unroll
        for (int nt = 0; nt < 4; ++nt) {
            v16bf b = sW[(kt * 4 + nt) * 32 + lane];  // one vector LDS load
            acc[nt] = __builtin_amdgcn_wmma_f32_16x16x32_bf16(
                false, a, false, b, (short)0, acc[nt], false, false);
        }
    }

#pragma unroll
    for (int nt = 0; nt < 4; ++nt) {
        const int col = nt * 16 + lrow;
        const float bb = bias[col];
#pragma unroll
        for (int v = 0; v < 8; ++v) {
            int row = m0 + v + (khalf ? 8 : 0);
            outp[(size_t)row * D + col] = leaky(acc[nt][v] + bb);
        }
    }
}

// MLP layer 0: h2 = leaky(dec(yenc) (Nx64) @ Wm0^T (64x128) + bm0)
__global__ void __launch_bounds__(256)
mlp_wmma(const unsigned* __restrict__ yenc, const v16bf* __restrict__ Wfrag,
         const float* __restrict__ bias, float* __restrict__ outp, int nrows) {
    __shared__ v16bf sW[512];  // 16 fragments x 32 lanes, 16KB
    stage_weights_16k(Wfrag, sW);

    const int wave = threadIdx.x >> 5;
    const int lane = threadIdx.x & 31;
    const int m0 = blockIdx.x * 128 + wave * 16;
    if (m0 >= nrows) return;

    const int lrow = lane & 15;
    const int khalf = lane >> 4;
    const int m = m0 + lrow;
    const unsigned* yrow = yenc + (size_t)m * D;

    v8f acc[8] = {};
#pragma unroll
    for (int kt = 0; kt < 2; ++kt) {
        const int kb = kt * 32 + khalf * 16;
        uint4 q0 = *(const uint4*)(yrow + kb);
        uint4 q1 = *(const uint4*)(yrow + kb + 4);
        uint4 q2 = *(const uint4*)(yrow + kb + 8);
        uint4 q3 = *(const uint4*)(yrow + kb + 12);
        v16bf a;
        a[0]  = (__bf16)dec_max(q0.x); a[1]  = (__bf16)dec_max(q0.y);
        a[2]  = (__bf16)dec_max(q0.z); a[3]  = (__bf16)dec_max(q0.w);
        a[4]  = (__bf16)dec_max(q1.x); a[5]  = (__bf16)dec_max(q1.y);
        a[6]  = (__bf16)dec_max(q1.z); a[7]  = (__bf16)dec_max(q1.w);
        a[8]  = (__bf16)dec_max(q2.x); a[9]  = (__bf16)dec_max(q2.y);
        a[10] = (__bf16)dec_max(q2.z); a[11] = (__bf16)dec_max(q2.w);
        a[12] = (__bf16)dec_max(q3.x); a[13] = (__bf16)dec_max(q3.y);
        a[14] = (__bf16)dec_max(q3.z); a[15] = (__bf16)dec_max(q3.w);
#pragma unroll
        for (int nt = 0; nt < 8; ++nt) {
            v16bf b = sW[(kt * 8 + nt) * 32 + lane];
            acc[nt] = __builtin_amdgcn_wmma_f32_16x16x32_bf16(
                false, a, false, b, (short)0, acc[nt], false, false);
        }
    }

#pragma unroll
    for (int nt = 0; nt < 8; ++nt) {
        const int col = nt * 16 + lrow;
        const float bb = bias[col];
#pragma unroll
        for (int v = 0; v < 8; ++v) {
            int row = m0 + v + (khalf ? 8 : 0);
            outp[(size_t)row * H + col] = leaky(acc[nt][v] + bb);
        }
    }
}

// out[n] = tanh(h2[n,:] . Wm1 + bm1), full f32.
__global__ void mlp_out(const float* __restrict__ h2, const float* __restrict__ Wm1,
                        const float* __restrict__ bm1, float* __restrict__ outp, int n) {
    int t = blockIdx.x * blockDim.x + threadIdx.x;
    if (t >= n) return;
    const float4* hp = (const float4*)(h2 + (size_t)t * H);
    const float4* wp = (const float4*)Wm1;
    float s = 0.0f;
#pragma unroll
    for (int i = 0; i < 32; ++i) {
        float4 a = hp[i], b = wp[i];
        s += a.x * b.x + a.y * b.y + a.z * b.z + a.w * b.w;
    }
    outp[t] = tanhf(s + bm1[0]);
}

static inline int cdiv(int a, int b) { return (a + b - 1) / b; }

extern "C" void kernel_launch(void* const* d_in, const int* in_sizes, int n_in,
                              void* d_out, int out_size, void* d_ws, size_t ws_size,
                              hipStream_t stream) {
    const float* X   = (const float*)d_in[0];
    const float* Ws0 = (const float*)d_in[1];
    const float* Wn0 = (const float*)d_in[2];
    const float* b0  = (const float*)d_in[3];
    const float* Ws1 = (const float*)d_in[4];
    const float* Wn1 = (const float*)d_in[5];
    const float* b1  = (const float*)d_in[6];
    const float* Wm0 = (const float*)d_in[7];
    const float* bm0 = (const float*)d_in[8];
    const float* Wm1 = (const float*)d_in[9];
    const float* bm1 = (const float*)d_in[10];
    const int* src  = (const int*)d_in[11];
    const int* dst  = (const int*)d_in[12];
    const int* csrc = (const int*)d_in[13];
    const int* cdst = (const int*)d_in[14];

    char* ws = (char*)d_ws;
    const size_t MB = 1ull << 20;
    float*    deg  = (float*)(ws + 0);                 // 0.4 MB
    float*    msum = (float*)(ws + 1 * MB);            // 25.6 MB
    float*    xx0  = (float*)(ws + 28 * MB);           // 25.6 MB
    float*    xx1  = (float*)(ws + 55 * MB);           // 25.6 MB
    unsigned* yenc = (unsigned*)(ws + 1 * MB);         // aliases msum (dead by pooling)
    float*    h2   = (float*)(ws + 28 * MB);           // aliases xx0 (dead by MLP)
    __bf16*   W0   = (__bf16*)(ws + 82 * MB);
    __bf16*   W1   = (__bf16*)(ws + 82 * MB + 32768);
    __bf16*   WM   = (__bf16*)(ws + 82 * MB + 65536);
    float*    outp = (float*)d_out;

    prep_weights<<<cdiv(24576, 256), 256, 0, stream>>>(Ws0, Wn0, Ws1, Wn1, Wm0, W0, W1, WM);

    // Layer 0
    zero_u32<<<cdiv(N_LV0, 256), 256, 0, stream>>>((uint32_t*)deg, N_LV0);
    zero_u32<<<cdiv(N_LV0 * D, 256), 256, 0, stream>>>((uint32_t*)msum, N_LV0 * D);
    degree_kernel<<<cdiv(E_G, 256), 256, 0, stream>>>(dst, deg, E_G);
    scatter_add<<<cdiv(2 * E_G, 256), 256, 0, stream>>>(X, src, dst, msum, E_G);
    sage_wmma<<<cdiv(N_LV0, 128), 256, 0, stream>>>(X, msum, deg, (const v16bf*)W0, b0, xx0, N_LV0);

    // Layer 1 (deg unchanged: it depends only on dst)
    zero_u32<<<cdiv(N_LV0 * D, 256), 256, 0, stream>>>((uint32_t*)msum, N_LV0 * D);
    scatter_add<<<cdiv(2 * E_G, 256), 256, 0, stream>>>(xx0, src, dst, msum, E_G);
    sage_wmma<<<cdiv(N_LV0, 128), 256, 0, stream>>>(xx0, msum, deg, (const v16bf*)W1, b1, xx1, N_LV0);

    // Pooling + MLP
    zero_u32<<<cdiv(N_NET * D, 256), 256, 0, stream>>>((uint32_t*)yenc, N_NET * D);
    pool_max<<<cdiv(2 * E_C, 256), 256, 0, stream>>>(xx1, csrc, cdst, yenc, E_C);
    mlp_wmma<<<cdiv(N_NET, 128), 256, 0, stream>>>(yenc, (const v16bf*)WM, bm0, h2, N_NET);
    mlp_out<<<cdiv(N_NET, 256), 256, 0, stream>>>(h2, Wm1, bm1, outp, N_NET);
}
